// EGNNLayerTSP_31653908971881
// MI455X (gfx1250) — matrix operand order repaired and checked by reading, not compiled, not run on hardware
//
#include <hip/hip_runtime.h>

// ---------------------------------------------------------------------------
// EGNN layer for MI455X (gfx1250): bf16 WMMA GEMMs, fused edge pipeline.
// B=16, N=2000, E=32000, NODE_DIM=128, EDGE_DIM=64, HIDDEN=256, COORD=2
// ---------------------------------------------------------------------------

#define BB 16
#define NN 2000
#define EE 32000
#define ND 128
#define ED 64
#define HH 256

typedef __attribute__((ext_vector_type(16))) __bf16 v16bf;
typedef __attribute__((ext_vector_type(8)))  float  v8f;

__device__ __forceinline__ unsigned short f2bf(float f) {
  union { float f; unsigned u; } x; x.f = f;
  unsigned r = x.u + 0x7FFFu + ((x.u >> 16) & 1u);   // round-to-nearest-even
  return (unsigned short)(r >> 16);
}
__device__ __forceinline__ float bf2f(unsigned short h) {
  union { unsigned u; float f; } x; x.u = ((unsigned)h) << 16;
  return x.f;
}
__device__ __forceinline__ float silu(float v) {
  return v / (1.f + __expf(-v));
}

// ---------------------------------------------------------------------------
// One wave computes a 16x16 output tile: C[mt,nt] = A(lds) x B(packed global).
// A: bf16 rows in LDS, stride strideA (multiple of 8).  A-fragment layout per
// ISA 7.12.2: lanes 0-15 -> K {0..7,16..23}, lanes 16-31 -> K {8..15,24..31};
// per lane that is two contiguous 16-byte chunks of its row.
// B: packed so lane reads 2 x uint4 at ((kt*NT+nt)*32+lane)*2.
// ---------------------------------------------------------------------------
__device__ __forceinline__ v8f gemm_tile(const unsigned short* ldsA, int strideA,
                                         const uint4* __restrict__ Wp,
                                         int ktiles, int NT, int mt, int nt, int lane) {
  v8f acc = {0.f,0.f,0.f,0.f,0.f,0.f,0.f,0.f};
  const int m     = lane & 15;
  const int khalf = lane >> 4;
  const unsigned short* rowp = ldsA + (mt * 16 + m) * strideA;
  for (int kt = 0; kt < ktiles; ++kt) {
    union { v16bf v; uint4 u[2]; } A, Bm;
    const unsigned short* ap = rowp + kt * 32 + 8 * khalf;
    A.u[0] = *(const uint4*)(ap);
    A.u[1] = *(const uint4*)(ap + 16);
    const uint4* bp = Wp + ((size_t)(kt * NT + nt) * 32 + lane) * 2;
    Bm.u[0] = bp[0];
    Bm.u[1] = bp[1];
    if (kt + 1 < ktiles)
      __builtin_prefetch(Wp + ((size_t)((kt + 1) * NT + nt) * 32 + lane) * 2, 0, 3);
    acc = __builtin_amdgcn_wmma_f32_16x16x32_bf16(
        /*neg_a=*/false, A.v, /*neg_b=*/false, Bm.v,
        /*c_mod=*/(short)0, acc, /*reuse_a=*/false, /*reuse_b=*/false);
  }
  return acc;
}

// D layout (ISA 7.12.2): VGPR r, lanes 0-15 -> (M=r, N=lane); lanes 16-31 -> (M=8+r).
__device__ __forceinline__ void store_tile(unsigned short* buf, int S, v8f acc,
                                           int mt, int nt, const float* bias,
                                           bool act, int lane) {
  const int n = lane & 15, m0 = (lane >> 4) * 8;
  const float bv = bias[nt * 16 + n];
#pragma unroll
  for (int r = 0; r < 8; ++r) {
    float v = acc[r] + bv;
    if (act) v = silu(v);
    buf[(mt * 16 + m0 + r) * S + nt * 16 + n] = f2bf(v);
  }
}

// Row-wise LayerNorm (in-place) over C cols of a 128-row bf16 LDS buffer.
// 256 threads, 2 per row.  s_part: 512 floats.
__device__ __forceinline__ void ln_rows_bf16(unsigned short* buf, int S, int C,
                                             const float* g, const float* bt,
                                             float* s_part) {
  const int t = threadIdx.x, row = t >> 1, half = t & 1, c0 = half * (C >> 1);
  float sum = 0.f, sq = 0.f;
  for (int k = 0; k < (C >> 1); ++k) {
    float v = bf2f(buf[row * S + c0 + k]);
    sum += v; sq += v * v;
  }
  s_part[t] = sum; s_part[256 + t] = sq;
  __syncthreads();
  float ts = s_part[row * 2] + s_part[row * 2 + 1];
  float tq = s_part[256 + row * 2] + s_part[256 + row * 2 + 1];
  float mu = ts / (float)C;
  float rstd = rsqrtf(tq / (float)C - mu * mu + 1e-5f);
  for (int k = 0; k < (C >> 1); ++k) {
    int c = c0 + k;
    float v = bf2f(buf[row * S + c]);
    buf[row * S + c] = f2bf((v - mu) * rstd * g[c] + bt[c]);
  }
  __syncthreads();
}

// ---------------------------------------------------------------------------
// Weight packer: fp32 (fin x fout) row-major -> bf16 WMMA-B layout.
// Block = one (kt, nt) tile; 256 threads = 32 lanes x 8 dwords.
// B layout: lane = (khalf<<4)|n ; dword j holds K = kb+2(j&3)+16(j>>2), +1 (kb = 8*khalf).
// ---------------------------------------------------------------------------
__global__ void pack_w(const float* __restrict__ W, unsigned* __restrict__ P,
                       int fin, int fout) {
  const int NT = fout >> 4;
  const int lane = threadIdx.x >> 3, j = threadIdx.x & 7;
  const int kt = blockIdx.x / NT, nt = blockIdx.x % NT;
  const int n  = nt * 16 + (lane & 15);
  const int k0 = kt * 32 + ((lane >> 4) << 3) + ((j >> 2) << 4) + 2 * (j & 3);
  float a = (k0     < fin) ? W[(size_t)k0 * fout + n]       : 0.f;
  float b = (k0 + 1 < fin) ? W[(size_t)(k0 + 1) * fout + n] : 0.f;
  P[((size_t)blockIdx.x * 32 + lane) * 8 + j] =
      (unsigned)f2bf(a) | ((unsigned)f2bf(b) << 16);
}

__global__ void init_kernel(const float* __restrict__ x, float* __restrict__ out_x,
                            float* __restrict__ hagg) {
  size_t i = (size_t)blockIdx.x * blockDim.x + threadIdx.x;
  const size_t total = (size_t)BB * NN * HH;
  const size_t step  = (size_t)gridDim.x * blockDim.x;
  for (; i < total; i += step) {
    hagg[i] = 0.f;
    if (i < (size_t)BB * NN * 2) out_x[i] = x[i];
  }
}

// ---------------------------------------------------------------------------
// Fused edge kernel: 128 edges / workgroup, 256 threads (8 waves).
// ---------------------------------------------------------------------------
struct EdgeArgs {
  const float *h, *x, *e; const int* ei;
  const uint4 *W0, *W1, *W2, *Wc0, *We0, *We1;
  const float *b0, *b1, *b2, *bc0, *wc1, *be0, *be1;
  const float *g_msg, *bt_msg, *g_eln, *bt_eln, *g_enorm, *bt_enorm;
  float *hagg, *out_x, *out_e;
};

#define SA 360   // msg_in stride (K padded to 352)
#define S1 264   // hidden stride
#define SE 328   // edge-concat stride (K=320)

__global__ __launch_bounds__(256) void egnn_edge_kernel(EdgeArgs a) {
  extern __shared__ char smem[];
  int*   s_row  = (int*)smem;
  int*   s_col  = s_row + 128;
  float* s_xd0  = (float*)(s_col + 128);
  float* s_xd1  = s_xd0 + 128;
  float* s_dist = s_xd1 + 128;
  float* s_part = s_dist + 128;                       // 512 floats
  unsigned short* bufA = (unsigned short*)(s_part + 512);  // 128*360 bf16
  unsigned short* buf1 = bufA + 128 * SA;                  // 128*264 bf16
  unsigned short* buf2 = buf1 + 128 * S1;                  // 128*264 bf16
  float* s_en = (float*)bufA;                              // reuse: 128*64 f32

  const int tid = threadIdx.x, lane = tid & 31, wave = tid >> 5;
  const int b     = blockIdx.x / (EE / 128);
  const int eBase = (blockIdx.x % (EE / 128)) * 128;

  // 1. indices + coordinate geometry
  if (tid < 128) {
    int r = a.ei[eBase + tid];
    int c = a.ei[EE + eBase + tid];
    s_row[tid] = r; s_col[tid] = c;
    const float* xb = a.x + (size_t)b * NN * 2;
    float dx = xb[c * 2 + 0] - xb[r * 2 + 0];
    float dy = xb[c * 2 + 1] - xb[r * 2 + 1];
    s_xd0[tid] = dx; s_xd1[tid] = dy;
    s_dist[tid] = sqrtf(dx * dx + dy * dy);
  }
  __syncthreads();

  // 2. stage msg_in = [h_row | h_col | dist | e | pad] as bf16 (K=352)
  {
    const float* hb = a.h + (size_t)b * NN * ND;
    const float* eb = a.e + (size_t)b * EE * ED;
    for (int idx = tid; idx < 128 * 352; idx += 256) {
      int el = idx / 352, k = idx - el * 352;
      float v;
      if      (k < 128) v = hb[s_row[el] * ND + k];
      else if (k < 256) v = hb[s_col[el] * ND + (k - 128)];
      else if (k == 256) v = s_dist[el];
      else if (k < 321) v = eb[(size_t)(eBase + el) * ED + (k - 257)];
      else v = 0.f;
      bufA[el * SA + k] = f2bf(v);
    }
  }
  __syncthreads();

  // 3. m = silu(msg0(msg_in))          (K=352 -> 11 ktiles)
  for (int nt = wave; nt < 16; nt += 8)
    for (int mt = 0; mt < 8; ++mt)
      store_tile(buf1, S1, gemm_tile(bufA, SA, a.W0, 11, 16, mt, nt, lane),
                 mt, nt, a.b0, true, lane);
  __syncthreads();

  // 4. m = layernorm(msg_ln, m)
  ln_rows_bf16(buf1, S1, HH, a.g_msg, a.bt_msg, s_part);

  // 5. m = silu(msg1(m))
  for (int nt = wave; nt < 16; nt += 8)
    for (int mt = 0; mt < 8; ++mt)
      store_tile(buf2, S1, gemm_tile(buf1, S1, a.W1, 8, 16, mt, nt, lane),
                 mt, nt, a.b1, true, lane);
  __syncthreads();

  // 6. m = msg2(m)  -> buf1 (bf16)  +  scatter-add into h_agg (fp32 atomics)
  for (int nt = wave; nt < 16; nt += 8)
    for (int mt = 0; mt < 8; ++mt) {
      v8f acc = gemm_tile(buf2, S1, a.W2, 8, 16, mt, nt, lane);
      const int n = lane & 15, m0 = (lane >> 4) * 8;
      const float bv = a.b2[nt * 16 + n];
#pragma unroll
      for (int r = 0; r < 8; ++r) {
        int mrow = mt * 16 + m0 + r;
        float v = acc[r] + bv;
        buf1[mrow * S1 + nt * 16 + n] = f2bf(v);
        atomicAdd(&a.hagg[((size_t)b * NN + s_row[mrow]) * HH + nt * 16 + n], v);
      }
    }
  __syncthreads();

  // 7. c1 = silu(coord0(m))
  for (int nt = wave; nt < 16; nt += 8)
    for (int mt = 0; mt < 8; ++mt)
      store_tile(buf2, S1, gemm_tile(buf1, S1, a.Wc0, 8, 16, mt, nt, lane),
                 mt, nt, a.bc0, true, lane);
  __syncthreads();

  // 8. cw = c1 . wc1 ; x_update scatter-add
  {
    const int row = tid >> 1, half = tid & 1;
    float s = 0.f;
    for (int k = 0; k < 128; ++k)
      s += bf2f(buf2[row * S1 + half * 128 + k]) * a.wc1[half * 128 + k];
    s_part[tid] = s;
    __syncthreads();
    if (half == 0) {
      float cw  = s_part[tid] + s_part[tid + 1];
      float inv = 1.f / (s_dist[row] + 1e-8f);
      float* xo = a.out_x + ((size_t)b * NN + s_row[row]) * 2;
      atomicAdd(xo + 0, cw * s_xd0[row] * inv);
      atomicAdd(xo + 1, cw * s_xd1[row] * inv);
    }
  }
  __syncthreads();

  // 9. stage edge concat [e | m]  (K=320)
  {
    const float* eb = a.e + (size_t)b * EE * ED;
    for (int idx = tid; idx < 128 * 320; idx += 256) {
      int el = idx / 320, k = idx - el * 320;
      bufA[el * SE + k] = (k < 64)
          ? f2bf(eb[(size_t)(eBase + el) * ED + k])
          : buf1[el * S1 + (k - 64)];
    }
  }
  __syncthreads();

  // 10. en = silu(edge0(cat))
  for (int nt = wave; nt < 16; nt += 8)
    for (int mt = 0; mt < 8; ++mt)
      store_tile(buf2, S1, gemm_tile(bufA, SE, a.We0, 10, 16, mt, nt, lane),
                 mt, nt, a.be0, true, lane);
  __syncthreads();

  // 11. en = layernorm(edge_ln, en)
  ln_rows_bf16(buf2, S1, HH, a.g_eln, a.bt_eln, s_part);

  // 12. en2 = edge1(en)  (NOUT=64 -> 4 n-tiles, waves 0..3)
  for (int nt = wave; nt < 4; nt += 8)
    for (int mt = 0; mt < 8; ++mt) {
      v8f acc = gemm_tile(buf2, S1, a.We1, 8, 4, mt, nt, lane);
      const int n = lane & 15, m0 = (lane >> 4) * 8;
      const float bv = a.be1[nt * 16 + n];
#pragma unroll
      for (int r = 0; r < 8; ++r)
        s_en[(mt * 16 + m0 + r) * 64 + nt * 16 + n] = acc[r] + bv;
    }
  __syncthreads();

  // 13. e_new = layernorm(edge_norm, e + en2)
  {
    const int row = tid >> 1, half = tid & 1, c0 = half * 32;
    const float* erow = a.e + ((size_t)b * EE + eBase + row) * ED;
    float sum = 0.f, sq = 0.f;
    for (int k = 0; k < 32; ++k) {
      float v = erow[c0 + k] + s_en[row * 64 + c0 + k];
      s_en[row * 64 + c0 + k] = v;
      sum += v; sq += v * v;
    }
    s_part[tid] = sum; s_part[256 + tid] = sq;
    __syncthreads();
    float ts = s_part[row * 2] + s_part[row * 2 + 1];
    float tq = s_part[256 + row * 2] + s_part[256 + row * 2 + 1];
    float mu = ts / 64.f, rstd = rsqrtf(tq / 64.f - mu * mu + 1e-5f);
    float* orow = a.out_e + ((size_t)b * EE + eBase + row) * ED;
    for (int k = 0; k < 32; ++k) {
      int c = c0 + k;
      orow[c] = (s_en[row * 64 + c] - mu) * rstd * a.g_enorm[c] + a.bt_enorm[c];
    }
  }
}

// ---------------------------------------------------------------------------
// Node kernel: 128 nodes / workgroup (16 tiles/batch, last tile ragged).
// ---------------------------------------------------------------------------
struct NodeArgs {
  const float *h, *hagg;
  const uint4 *Wn0, *Wn1;
  const float *bn0, *bn1, *g_nln, *bt_nln, *g_nnorm, *bt_nnorm;
  float* out_h;
};

#define SN 392   // concat stride (K=384)

__global__ __launch_bounds__(256) void egnn_node_kernel(NodeArgs a) {
  extern __shared__ char smem[];
  float* s_part = (float*)smem;                              // 512 floats
  unsigned short* bufA = (unsigned short*)(smem + 2048);     // 128*392 bf16
  unsigned short* buf1 = bufA + 128 * SN;                    // 128*264 bf16
  float* s_hn = (float*)bufA;                                // reuse: 128*128 f32

  const int tid = threadIdx.x, lane = tid & 31, wave = tid >> 5;
  const int b = blockIdx.x >> 4;
  const int nBase = (blockIdx.x & 15) * 128;

  // stage [h | h_agg] (K=384), zeros past N
  for (int idx = tid; idx < 128 * 384; idx += 256) {
    int rl = idx / 384, k = idx - rl * 384;
    int node = nBase + rl;
    float v = 0.f;
    if (node < NN)
      v = (k < 128) ? a.h[((size_t)b * NN + node) * ND + k]
                    : a.hagg[((size_t)b * NN + node) * HH + (k - 128)];
    bufA[rl * SN + k] = f2bf(v);
  }
  __syncthreads();

  // hn = silu(node0(cat))  (12 ktiles)
  for (int nt = wave; nt < 16; nt += 8)
    for (int mt = 0; mt < 8; ++mt)
      store_tile(buf1, S1, gemm_tile(bufA, SN, a.Wn0, 12, 16, mt, nt, lane),
                 mt, nt, a.bn0, true, lane);
  __syncthreads();

  // hn = layernorm(node_ln, hn)
  ln_rows_bf16(buf1, S1, HH, a.g_nln, a.bt_nln, s_part);

  // hn2 = node1(hn)  (NOUT=128 -> 8 n-tiles, one per wave)
  for (int nt = wave; nt < 8; nt += 8)
    for (int mt = 0; mt < 8; ++mt) {
      v8f acc = gemm_tile(buf1, S1, a.Wn1, 8, 8, mt, nt, lane);
      const int n = lane & 15, m0 = (lane >> 4) * 8;
      const float bv = a.bn1[nt * 16 + n];
#pragma unroll
      for (int r = 0; r < 8; ++r)
        s_hn[(mt * 16 + m0 + r) * 128 + nt * 16 + n] = acc[r] + bv;
    }
  __syncthreads();

  // h_new = layernorm(node_norm, h + hn2)
  {
    const int row = tid >> 1, half = tid & 1, c0 = half * 64;
    const int node = nBase + row;
    float sum = 0.f, sq = 0.f;
    if (node < NN) {
      const float* hrow = a.h + ((size_t)b * NN + node) * ND;
      for (int k = 0; k < 64; ++k) {
        float v = hrow[c0 + k] + s_hn[row * 128 + c0 + k];
        s_hn[row * 128 + c0 + k] = v;
        sum += v; sq += v * v;
      }
    }
    s_part[tid] = sum; s_part[256 + tid] = sq;
    __syncthreads();
    if (node < NN) {
      float ts = s_part[row * 2] + s_part[row * 2 + 1];
      float tq = s_part[256 + row * 2] + s_part[256 + row * 2 + 1];
      float mu = ts / 128.f, rstd = rsqrtf(tq / 128.f - mu * mu + 1e-5f);
      float* orow = a.out_h + ((size_t)b * NN + node) * ND;
      for (int k = 0; k < 64; ++k) {
        int c = c0 + k;
        orow[c] = (s_hn[row * 128 + c] - mu) * rstd * a.g_nnorm[c] + a.bt_nnorm[c];
      }
    }
  }
}

// ---------------------------------------------------------------------------
extern "C" void kernel_launch(void* const* d_in, const int* in_sizes, int n_in,
                              void* d_out, int out_size, void* d_ws, size_t ws_size,
                              hipStream_t stream) {
  (void)in_sizes; (void)n_in; (void)out_size; (void)ws_size;
  const float* h   = (const float*)d_in[0];
  const float* x   = (const float*)d_in[1];
  const float* e   = (const float*)d_in[2];
  const int*   ei  = (const int*)d_in[3];
  const float* W0  = (const float*)d_in[4];  const float* b0  = (const float*)d_in[5];
  const float* gm  = (const float*)d_in[6];  const float* btm = (const float*)d_in[7];
  const float* W1  = (const float*)d_in[8];  const float* b1  = (const float*)d_in[9];
  const float* W2  = (const float*)d_in[10]; const float* b2  = (const float*)d_in[11];
  const float* Wc0 = (const float*)d_in[12]; const float* bc0 = (const float*)d_in[13];
  const float* wc1 = (const float*)d_in[14];
  const float* Wn0 = (const float*)d_in[15]; const float* bn0 = (const float*)d_in[16];
  const float* gnl = (const float*)d_in[17]; const float* btnl= (const float*)d_in[18];
  const float* Wn1 = (const float*)d_in[19]; const float* bn1 = (const float*)d_in[20];
  const float* We0 = (const float*)d_in[21]; const float* be0 = (const float*)d_in[22];
  const float* gel = (const float*)d_in[23]; const float* btel= (const float*)d_in[24];
  const float* We1 = (const float*)d_in[25]; const float* be1 = (const float*)d_in[26];
  const float* gnn = (const float*)d_in[27]; const float* btnn= (const float*)d_in[28];
  const float* gen = (const float*)d_in[29]; const float* bten= (const float*)d_in[30];

  float* out_h = (float*)d_out;
  float* out_x = out_h + (size_t)BB * NN * ND;
  float* out_e = out_x + (size_t)BB * NN * 2;

  // workspace carve-up
  char* ws = (char*)d_ws;
  float* hagg = (float*)ws;
  size_t off = (((size_t)BB * NN * HH * 4) + 255) & ~(size_t)255;
  auto take = [&](size_t bytes) -> void* {
    void* p = ws + off;
    off += (bytes + 255) & ~(size_t)255;
    return p;
  };
  unsigned* W0p  = (unsigned*)take((size_t)352 * 256 * 2);
  unsigned* W1p  = (unsigned*)take((size_t)256 * 256 * 2);
  unsigned* W2p  = (unsigned*)take((size_t)256 * 256 * 2);
  unsigned* Wc0p = (unsigned*)take((size_t)256 * 256 * 2);
  unsigned* Wn0p = (unsigned*)take((size_t)384 * 256 * 2);
  unsigned* Wn1p = (unsigned*)take((size_t)256 * 128 * 2);
  unsigned* We0p = (unsigned*)take((size_t)320 * 256 * 2);
  unsigned* We1p = (unsigned*)take((size_t)256 * 64 * 2);

  // pack weights into WMMA-B layout (bf16)
  pack_w<<<11 * 16, 256, 0, stream>>>(W0,  W0p,  321, 256);
  pack_w<<< 8 * 16, 256, 0, stream>>>(W1,  W1p,  256, 256);
  pack_w<<< 8 * 16, 256, 0, stream>>>(W2,  W2p,  256, 256);
  pack_w<<< 8 * 16, 256, 0, stream>>>(Wc0, Wc0p, 256, 256);
  pack_w<<<12 * 16, 256, 0, stream>>>(Wn0, Wn0p, 384, 256);
  pack_w<<< 8 *  8, 256, 0, stream>>>(Wn1, Wn1p, 256, 128);
  pack_w<<<10 * 16, 256, 0, stream>>>(We0, We0p, 320, 256);
  pack_w<<< 8 *  4, 256, 0, stream>>>(We1, We1p, 256,  64);

  init_kernel<<<2048, 256, 0, stream>>>(x, out_x, hagg);

  EdgeArgs ea;
  ea.h = h; ea.x = x; ea.e = e; ea.ei = ei;
  ea.W0 = (const uint4*)W0p; ea.W1 = (const uint4*)W1p; ea.W2 = (const uint4*)W2p;
  ea.Wc0 = (const uint4*)Wc0p; ea.We0 = (const uint4*)We0p; ea.We1 = (const uint4*)We1p;
  ea.b0 = b0; ea.b1 = b1; ea.b2 = b2; ea.bc0 = bc0; ea.wc1 = wc1;
  ea.be0 = be0; ea.be1 = be1;
  ea.g_msg = gm; ea.bt_msg = btm; ea.g_eln = gel; ea.bt_eln = btel;
  ea.g_enorm = gen; ea.bt_enorm = bten;
  ea.hagg = hagg; ea.out_x = out_x; ea.out_e = out_e;

  NodeArgs na;
  na.h = h; na.hagg = hagg;
  na.Wn0 = (const uint4*)Wn0p; na.Wn1 = (const uint4*)Wn1p;
  na.bn0 = bn0; na.bn1 = bn1;
  na.g_nln = gnl; na.bt_nln = btnl; na.g_nnorm = gnn; na.bt_nnorm = btnn;
  na.out_h = out_h;

  const size_t edgeSmem = 4608 + (size_t)128 * SA * 2 + 2 * (size_t)128 * S1 * 2; // 231936
  const size_t nodeSmem = 2048 + (size_t)128 * SN * 2 + (size_t)128 * S1 * 2;     // 169984
  hipFuncSetAttribute(reinterpret_cast<const void*>(egnn_edge_kernel),
                      hipFuncAttributeMaxDynamicSharedMemorySize, (int)edgeSmem);
  hipFuncSetAttribute(reinterpret_cast<const void*>(egnn_node_kernel),
                      hipFuncAttributeMaxDynamicSharedMemorySize, (int)nodeSmem);

  egnn_edge_kernel<<<BB * (EE / 128), 256, edgeSmem, stream>>>(ea);
  egnn_node_kernel<<<BB * 16, 256, nodeSmem, stream>>>(na);
}
